// PatchConv2d_65025804861587
// MI455X (gfx1250) — matrix operand order repaired and checked by reading, not compiled, not run on hardware
//
#include <hip/hip_runtime.h>

// PatchConv2d fused implicit-GEMM for gfx1250 (MI455X, wave32, bf16x3 WMMA).
//
// out[b,o,h,w] = conv3x3(x,W)[b,o,h,w] + 0.1*(h+w)*patch_sum[b,h,w] + bias[o]
//   grid2d[h,w] == h + w (Ho==Wo==128)
//   patch_sum   == exact fp32 column sums of the im2col B matrix, computed
//                  during staging (not in the hot loop).
//
// GEMM: M=128 (8 waves x 16 O), N=16 spatial, K=64*9=576 in 18 blocks of 32.
// fp32 operands are split x = hi + lo (bf16 each); per K-block we issue
//   acc += Ahi*Bhi + Ahi*Blo + Alo*Bhi   (v_wmma_f32_16x16x32_bf16, fp32 acc)
// giving ~2^-16 relative accuracy at 8x the K-depth per WMMA issue vs the
// fp32 16x16x4 path -> memory-roofline bound (200 MB @ 23.3 TB/s ~ 8.6 us).

typedef __attribute__((ext_vector_type(16))) __bf16        v16bf;
typedef __attribute__((ext_vector_type(8)))  float          v8f;
typedef __attribute__((ext_vector_type(8)))  unsigned short v8us;

#define CIN   64
#define COUT  128
#define HW    128
#define KTOT  576                 // 64 * 9
#define NKB   18                  // K blocks of 32
#define BCOEF 0.1f
#define SBS   584                 // LDS B row stride: 292 dwords -> conflict-free

static __device__ __forceinline__ unsigned short f2bf(float f) {  // RNE
    unsigned u = __float_as_uint(f);
    u += 0x7FFFu + ((u >> 16) & 1u);
    return (unsigned short)(u >> 16);
}
static __device__ __forceinline__ float bf2f(unsigned short h) {
    return __uint_as_float((unsigned)h << 16);
}

// ---- One-shot weight split into WMMA-native swizzled hi/lo bf16 arrays.
// Dest index j = ((ot*NKB + t)*32 + lane)*16 + e ; lane = g*16 + m.
// e 0..7  -> K = t*32 + g*8 + e        (A VGPR0-3 half)
// e 8..15 -> K = t*32 + 16 + g*8 + e-8 (A VGPR4-7 half)
__global__ __launch_bounds__(256) void
split_weights(const float* __restrict__ w,
              unsigned short* __restrict__ ahi,
              unsigned short* __restrict__ alo) {
    int j = blockIdx.x * 256 + threadIdx.x;
    if (j >= COUT * KTOT) return;
    int e    = j & 15;
    int lane = (j >> 4) & 31;
    int blk  = j >> 9;                       // ot*NKB + t
    int t    = blk % NKB;
    int ot   = blk / NKB;
    int m = lane & 15, g = lane >> 4;
    int k = t * 32 + ((e < 8) ? (g * 8 + e) : (16 + g * 8 + (e - 8)));
    int o = (ot << 4) + m;
    float v = w[o * KTOT + k];
    unsigned short h = f2bf(v);
    ahi[j] = h;
    alo[j] = f2bf(v - bf2f(h));
}

__global__ __launch_bounds__(256) void
patchconv_wmma_bf16x3(const float* __restrict__ x,
                      const unsigned short* __restrict__ ahi,
                      const unsigned short* __restrict__ alo,
                      const float* __restrict__ bias,
                      float* __restrict__ out) {
    __shared__ unsigned short bhi[16 * SBS];   // 18.25 KB
    __shared__ unsigned short blo[16 * SBS];   // 18.25 KB
    __shared__ float psum_arr[16 * 16];        // 1 KB (patch_sum partials)

    // Block -> (batch, row h, 16-wide w tile). 16*128*8 = 16384 blocks.
    const int tile = blockIdx.x;
    const int b    = tile >> 10;
    const int sp   = tile & 1023;
    const int h    = sp >> 3;
    const int w0   = (sp & 7) << 4;
    const int tid  = threadIdx.x;

    // ---- Stage: im2col-expand + hi/lo split into LDS; exact fp32 patch_sum
    // partials on the side (every i of this thread has n == tid & 15).
    const float* xb = x + (size_t)b * CIN * HW * HW;
    float ps = 0.0f;
    for (int i = tid; i < KTOT * 16; i += 256) {
        int k   = i >> 4;
        int n   = i & 15;
        int ci  = k / 9;
        int rem = k - ci * 9;
        int kh  = rem / 3;
        int kw  = rem - kh * 3;
        int hh  = h  + kh - 1;
        int ww  = w0 + n + kw - 1;
        float v = 0.0f;
        if ((unsigned)hh < HW && (unsigned)ww < HW)
            v = xb[(ci * HW + hh) * HW + ww];   // consecutive lanes: coalesced
        unsigned short hv = f2bf(v);
        bhi[n * SBS + k] = hv;
        blo[n * SBS + k] = f2bf(v - bf2f(hv));
        ps += v;
    }
    psum_arr[((tid >> 4) << 4) + (tid & 15)] = ps;
    __syncthreads();

    // ---- Wave-level 16x16 tile: wave i owns output channels 16i..16i+15.
    const int wave = tid >> 5;
    const int lane = tid & 31;
    const int m    = lane & 15;    // A row / B column / D column
    const int g    = lane >> 4;    // K-half selector (16-bit WMMA layout)

    // A: swizzled hi/lo, one contiguous 32B per lane per K-block (coalesced).
    const v8us* Ah = (const v8us*)ahi + ((size_t)wave * NKB * 32 + lane) * 2;
    const v8us* Al = (const v8us*)alo + ((size_t)wave * NKB * 32 + lane) * 2;
    // B: lanes 0-15 need K 0..7,16..23, lanes 16-31 need K 8..15,24..31 of each block.
    const unsigned short* Bh = &bhi[m * SBS + (g << 3)];
    const unsigned short* Bl = &blo[m * SBS + (g << 3)];

    __builtin_prefetch(Ah, 0, 0);              // global_prefetch_b8

    union U { v16bf v; v8us h[2]; };
    v8f acc = {};

    #pragma unroll 6
    for (int t = 0; t < NKB; ++t) {
        U Av_hi, Av_lo, Bv_hi, Bv_lo;
        Av_hi.h[0] = Ah[t * 64 + 0];                      // +1 KB/iter, coalesced
        Av_hi.h[1] = Ah[t * 64 + 1];
        Av_lo.h[0] = Al[t * 64 + 0];
        Av_lo.h[1] = Al[t * 64 + 1];
        Bv_hi.h[0] = *(const v8us*)(Bh + t * 32);         // ds_load_b128 x2
        Bv_hi.h[1] = *(const v8us*)(Bh + t * 32 + 16);
        Bv_lo.h[0] = *(const v8us*)(Bl + t * 32);
        Bv_lo.h[1] = *(const v8us*)(Bl + t * 32 + 16);

        acc = __builtin_amdgcn_wmma_f32_16x16x32_bf16(
            false, Av_hi.v, false, Bv_hi.v, (short)0, acc, false, false);
        acc = __builtin_amdgcn_wmma_f32_16x16x32_bf16(
            false, Av_hi.v, false, Bv_lo.v, (short)0, acc, false, false);
        acc = __builtin_amdgcn_wmma_f32_16x16x32_bf16(
            false, Av_lo.v, false, Bv_hi.v, (short)0, acc, false, false);
    }

    // ---- Epilogue: exact patch_sum column total + grid/bias fusion.
    float colsum = 0.0f;
    #pragma unroll
    for (int j = 0; j < 16; ++j) colsum += psum_arr[(j << 4) + m];
    const float gterm = BCOEF * (float)(h + w0 + m) * colsum;

    // C/D layout: VGPR r holds M=r (lanes 0-15) or M=r+8 (lanes 16-31).
    const int o0    = wave << 4;
    const int mbase = g << 3;
    float* ob = out + (((size_t)b * COUT + o0) * HW + h) * HW + w0 + m;
    #pragma unroll
    for (int r = 0; r < 8; ++r) {
        int M = r + mbase;
        ob[(size_t)M * HW * HW] = acc[r] + gterm + bias[o0 + M];
    }
}

extern "C" void kernel_launch(void* const* d_in, const int* in_sizes, int n_in,
                              void* d_out, int out_size, void* d_ws, size_t ws_size,
                              hipStream_t stream) {
    (void)in_sizes; (void)n_in; (void)ws_size; (void)out_size;
    const float* x    = (const float*)d_in[0];   // (16, 64, 128, 128)
    const float* wgt  = (const float*)d_in[1];   // (128, 64, 3, 3)
    const float* bias = (const float*)d_in[2];   // (128,)
    float*       out  = (float*)d_out;           // (16, 128, 128, 128)

    // Workspace: swizzled bf16 weight hi/lo, 2 * 73728 * 2 B = 294,912 B.
    unsigned short* ahi = (unsigned short*)d_ws;
    unsigned short* alo = ahi + COUT * KTOT;

    split_weights<<<(COUT * KTOT + 255) / 256, 256, 0, stream>>>(wgt, ahi, alo);
    patchconv_wmma_bf16x3<<<16 * 128 * 8, 256, 0, stream>>>(x, ahi, alo, bias, out);
}
